// AttSeqDecoder_54159537603049
// MI455X (gfx1250) — matrix operand order repaired and checked by reading, not compiled, not run on hardware
//
#include <hip/hip_runtime.h>
#include <hip/hip_bf16.h>

// ---------------------------------------------------------------------------
// AttSeqDecoder for MI455X (gfx1250): bf16 WMMA GEMMs + VALU attention.
// B=32, T=64, S=128, E=256, H=512, ENC=512, V=32000.
// Dominant work: per-step [32,512]x[512,32000] vocab projection -> wmma bf16.
// ---------------------------------------------------------------------------

typedef __bf16 bf16_t;
typedef __attribute__((ext_vector_type(16))) __bf16        v16bf;
typedef __attribute__((ext_vector_type(8)))  float         v8f;
typedef __attribute__((ext_vector_type(4)))  unsigned int  v4u;
typedef __attribute__((ext_vector_type(8)))  unsigned int  v8u;

#define DB   32      // batch
#define DT   64      // decode steps
#define DS_  128     // source length
#define DE   256     // embedding
#define DH   512     // hidden
#define DENC 512     // encoder size
#define DV   32000   // vocab

// ---------------------------------------------------------------------------
// WMMA fragment loaders (gfx1250 wave32; A and B both use contiguous k = 16*(lane/16) + s)
// ---------------------------------------------------------------------------

// A: 16x32 bf16, row-major source with leading dim lda (elements).
// lane<16 : row M=lane,    K in {0..15}
// lane>=16: row M=lane-16, K in {16..31}
__device__ __forceinline__ v16bf load_frag_a(const bf16_t* __restrict__ A, int lda,
                                             int mBase, int k0, int lane) {
  const bf16_t* p = A + (long long)(mBase + (lane & 15)) * lda + k0 + ((lane >> 4) << 4);
  return *(const v16bf*)p;          // 32B contiguous
}

// B: 32x16 bf16 where B[k][n] = W[n][k], W row-major [N, ldw].
// lane<16 : col N=lane,    K = k0+0..15   -> contiguous 16 bf16 of row n
// lane>=16: col N=lane-16, K = k0+16..31
__device__ __forceinline__ v16bf load_frag_b(const bf16_t* __restrict__ W, int ldw,
                                             int nBase, int k0, int lane) {
  const bf16_t* p = W + (long long)(nBase + (lane & 15)) * ldw + k0 + ((lane >> 4) << 4);
  return *(const v16bf*)p;          // 32B contiguous
}

// ---------------------------------------------------------------------------
// Generic WMMA GEMM: out[M,N] = A[M,K] @ W[N,K]^T (+ A2[M,K2] @ W2[N,K2]^T) + bias
// Block = 256 threads = 8 waves; wave tile = 16x16; block tile = 32(M) x 64(N).
// grid.x = N/64, grid.y = M/32. All call sites divide exactly (no guards, so
// EXEC is all-ones for every v_wmma as the ISA requires).
// ---------------------------------------------------------------------------
template <typename OutT>
__global__ void wmma_gemm_kernel(const bf16_t* __restrict__ A, int lda,
                                 const bf16_t* __restrict__ W, int ldw, int K,
                                 const bf16_t* __restrict__ A2, int lda2,
                                 const bf16_t* __restrict__ W2, int ldw2, int K2,
                                 const float* __restrict__ bias,
                                 OutT* __restrict__ out, long long ld_out) {
  const int lane  = threadIdx.x & 31;
  const int wave  = threadIdx.x >> 5;
  const int mBase = blockIdx.y * 32 + (wave >> 2) * 16;
  const int nBase = blockIdx.x * 64 + (wave & 3) * 16;

  v8f acc = {};
  for (int k0 = 0; k0 < K; k0 += 32) {
    v16bf a = load_frag_a(A, lda, mBase, k0, lane);
    v16bf b = load_frag_b(W, ldw, nBase, k0, lane);
    acc = __builtin_amdgcn_wmma_f32_16x16x32_bf16(false, a, false, b,
                                                  (short)0, acc, false, false);
  }
  if (A2) {  // fused second GEMM (uniform branch: EXEC unchanged)
    for (int k0 = 0; k0 < K2; k0 += 32) {
      v16bf a = load_frag_a(A2, lda2, mBase, k0, lane);
      v16bf b = load_frag_b(W2, ldw2, nBase, k0, lane);
      acc = __builtin_amdgcn_wmma_f32_16x16x32_bf16(false, a, false, b,
                                                    (short)0, acc, false, false);
    }
  }

  // D layout: VGPR e -> (m = mBase + 8*(lane>=16) + e, n = nBase + lane%16)
  const int   n    = nBase + (lane & 15);
  const float bv   = bias ? bias[n] : 0.0f;
  const int   mOff = (lane >> 4) << 3;
#pragma unroll
  for (int e = 0; e < 8; ++e) {
    long long m = mBase + mOff + e;
    out[m * ld_out + n] = (OutT)(acc[e] + bv);
  }
}

// ---------------------------------------------------------------------------
// Elementwise / setup kernels
// ---------------------------------------------------------------------------
__global__ void cvt_bf16_kernel(const float* __restrict__ src,
                                bf16_t* __restrict__ dst, long long n) {
  long long i = (long long)blockIdx.x * blockDim.x + threadIdx.x;
  if (i < n) dst[i] = (bf16_t)src[i];
}

__global__ void gather_embed_kernel(const float* __restrict__ table,
                                    const int* __restrict__ tok,
                                    bf16_t* __restrict__ dst, int rows, int E) {
  long long i = (long long)blockIdx.x * blockDim.x + threadIdx.x;
  if (i < (long long)rows * E) {
    int r = (int)(i / E), e = (int)(i % E);
    dst[i] = (bf16_t)table[(long long)tok[r] * E + e];
  }
}

__global__ void init_hc_kernel(const float* __restrict__ h, bf16_t* __restrict__ hc) {
  int i = blockIdx.x * blockDim.x + threadIdx.x;   // over B*H
  if (i < DB * DH) {
    int b = i >> 9, j = i & (DH - 1);
    hc[b * (DH + DENC) + j] = (bf16_t)h[i];
  }
}

__device__ __forceinline__ float sigmoidf_(float x) { return 1.0f / (1.0f + __expf(-x)); }

// gates [B,4H] f32 -> c,h f32 (state) + h as bf16 into hc_bf[:, 0:H]
__global__ void lstm_cell_kernel(const float* __restrict__ gates,
                                 const float* __restrict__ b_ih,
                                 const float* __restrict__ b_hh,
                                 float* __restrict__ c, float* __restrict__ h,
                                 bf16_t* __restrict__ hc_bf) {
  int i = blockIdx.x * blockDim.x + threadIdx.x;   // over B*H
  if (i >= DB * DH) return;
  int b = i >> 9, j = i & (DH - 1);
  const float* g = gates + (long long)b * 4 * DH;
  float xi = g[j]            + b_ih[j]            + b_hh[j];
  float xf = g[DH + j]       + b_ih[DH + j]       + b_hh[DH + j];
  float xg = g[2 * DH + j]   + b_ih[2 * DH + j]   + b_hh[2 * DH + j];
  float xo = g[3 * DH + j]   + b_ih[3 * DH + j]   + b_hh[3 * DH + j];
  float cn = sigmoidf_(xf) * c[i] + sigmoidf_(xi) * tanhf(xg);
  float hn = sigmoidf_(xo) * tanhf(cn);
  c[i] = cn;
  h[i] = hn;
  hc_bf[b * (DH + DENC) + j] = (bf16_t)hn;
}

// One block per batch row: scores = h . proj_enc, masked softmax, ctx = att@enc.
// ctx written as bf16 into hc_bf[:, H:H+ENC] for the fused Wc GEMM.
__global__ void attention_kernel(const float* __restrict__ h,
                                 const float* __restrict__ proj,   // [B,S,H] f32
                                 const float* __restrict__ enc,    // [B,S,ENC] f32
                                 const int* __restrict__ lens,
                                 bf16_t* __restrict__ hc_bf) {
  const int b = blockIdx.x, tid = threadIdx.x;
  const int lane = tid & 31, wave = tid >> 5;
  __shared__ float s_h[DH];
  __shared__ float s_sc[DS_];
  __shared__ float s_red[DS_];

  for (int i = tid; i < DH; i += 256) s_h[i] = h[b * DH + i];
  __syncthreads();

  const int len = lens[b];
  for (int s = wave; s < DS_; s += 8) {
    const float* pr = proj + ((long long)b * DS_ + s) * DH;
    float part = 0.f;
    for (int k = lane; k < DH; k += 32) part += s_h[k] * pr[k];
    for (int off = 16; off > 0; off >>= 1) part += __shfl_xor(part, off, 32);
    if (lane == 0) s_sc[s] = (s < len) ? part : -1e9f;
  }
  __syncthreads();

  // block softmax over S=128
  if (tid < DS_) s_red[tid] = s_sc[tid];
  __syncthreads();
  for (int off = 64; off > 0; off >>= 1) {
    if (tid < off) s_red[tid] = fmaxf(s_red[tid], s_red[tid + off]);
    __syncthreads();
  }
  const float mx = s_red[0];
  __syncthreads();
  if (tid < DS_) { float e = __expf(s_sc[tid] - mx); s_sc[tid] = e; s_red[tid] = e; }
  __syncthreads();
  for (int off = 64; off > 0; off >>= 1) {
    if (tid < off) s_red[tid] += s_red[tid + off];
    __syncthreads();
  }
  const float inv = 1.0f / s_red[0];
  __syncthreads();
  if (tid < DS_) s_sc[tid] *= inv;
  __syncthreads();

  // ctx: each thread accumulates 2 of the 512 encoder channels
  const int e0 = tid * 2;
  float a0 = 0.f, a1 = 0.f;
  for (int s = 0; s < DS_; ++s) {
    float w = s_sc[s];
    const float* pe = enc + ((long long)b * DS_ + s) * DENC + e0;
    a0 += w * pe[0];
    a1 += w * pe[1];
  }
  hc_bf[b * (DH + DENC) + DH + e0]     = (bf16_t)a0;
  hc_bf[b * (DH + DENC) + DH + e0 + 1] = (bf16_t)a1;
}

// ---------------------------------------------------------------------------
// Host orchestration
// ---------------------------------------------------------------------------
extern "C" void kernel_launch(void* const* d_in, const int* in_sizes, int n_in,
                              void* d_out, int out_size, void* d_ws, size_t ws_size,
                              hipStream_t stream) {
  (void)in_sizes; (void)n_in; (void)out_size; (void)ws_size;

  const int*   gold_tokens = (const int*)  d_in[0];   // [B,T]
  const float* enc_outputs = (const float*)d_in[1];   // [B,S,ENC]
  const int*   enc_lengths = (const int*)  d_in[2];   // [B]
  const float* h0          = (const float*)d_in[3];   // [B,H]
  const float* c0          = (const float*)d_in[4];   // [B,H]
  const float* embed_table = (const float*)d_in[5];   // [V,E]
  const float* W_ih        = (const float*)d_in[6];   // [4H,E]
  const float* W_hh        = (const float*)d_in[7];   // [4H,H]
  const float* b_ih        = (const float*)d_in[8];   // [4H]
  const float* b_hh        = (const float*)d_in[9];   // [4H]
  const float* Wa          = (const float*)d_in[10];  // [H,ENC]
  const float* Wc_w        = (const float*)d_in[11];  // [H, H+ENC]
  const float* Wc_b        = (const float*)d_in[12];  // [H]
  const float* Wo_w        = (const float*)d_in[13];  // [V,H]
  const float* Wo_b        = (const float*)d_in[14];  // [V]
  float* out = (float*)d_out;                          // [B,T,V]

  // --- workspace carve-out (256B aligned) ---
  char* w = (char*)d_ws;
  auto carve = [&](size_t bytes) { char* p = w; w += (bytes + 255) & ~(size_t)255; return p; };
  bf16_t* Wo_bf   = (bf16_t*)carve((size_t)DV * DH * 2);            // 32 MB (fits L2)
  bf16_t* Wih_bf  = (bf16_t*)carve((size_t)4 * DH * DE * 2);
  bf16_t* Whh_bf  = (bf16_t*)carve((size_t)4 * DH * DH * 2);
  bf16_t* Wc_bf   = (bf16_t*)carve((size_t)DH * (DH + DENC) * 2);
  bf16_t* Wa_bf   = (bf16_t*)carve((size_t)DH * DENC * 2);
  bf16_t* enc_bf  = (bf16_t*)carve((size_t)DB * DS_ * DENC * 2);
  bf16_t* embs_bf = (bf16_t*)carve((size_t)DB * DT * DE * 2);
  float*  proj    = (float*) carve((size_t)DB * DS_ * DH * 4);      // Wa @ enc
  float*  gates   = (float*) carve((size_t)DB * 4 * DH * 4);
  float*  hbuf    = (float*) carve((size_t)DB * DH * 4);
  float*  cbuf    = (float*) carve((size_t)DB * DH * 4);
  bf16_t* hc_bf   = (bf16_t*)carve((size_t)DB * (DH + DENC) * 2);   // [h | ctx] bf16
  bf16_t* ho_bf   = (bf16_t*)carve((size_t)DB * DH * 2);

  // --- one-time (per launch) conversions & precompute ---
  auto cvt = [&](const float* s, bf16_t* d, long long n) {
    cvt_bf16_kernel<<<(unsigned)((n + 255) / 256), 256, 0, stream>>>(s, d, n);
  };
  cvt(W_ih, Wih_bf, (long long)4 * DH * DE);
  cvt(W_hh, Whh_bf, (long long)4 * DH * DH);
  cvt(Wc_w, Wc_bf,  (long long)DH * (DH + DENC));
  cvt(Wo_w, Wo_bf,  (long long)DV * DH);
  cvt(Wa,   Wa_bf,  (long long)DH * DENC);
  cvt(enc_outputs, enc_bf, (long long)DB * DS_ * DENC);
  gather_embed_kernel<<<(DB * DT * DE + 255) / 256, 256, 0, stream>>>(
      embed_table, gold_tokens, embs_bf, DB * DT, DE);

  hipMemcpyAsync(hbuf, h0, (size_t)DB * DH * 4, hipMemcpyDeviceToDevice, stream);
  hipMemcpyAsync(cbuf, c0, (size_t)DB * DH * 4, hipMemcpyDeviceToDevice, stream);
  init_hc_kernel<<<(DB * DH + 255) / 256, 256, 0, stream>>>(hbuf, hc_bf);

  // proj_enc[b,s,:] = enc[b,s,:] @ Wa^T   -> GEMM M=B*S=4096, N=H=512, K=ENC=512
  wmma_gemm_kernel<float><<<dim3(DH / 64, (DB * DS_) / 32), 256, 0, stream>>>(
      enc_bf, DENC, Wa_bf, DENC, DENC,
      nullptr, 0, nullptr, 0, 0, nullptr, proj, (long long)DH);

  // --- sequential decode (recurrence over T) ---
  for (int t = 0; t < DT; ++t) {
    // gates = e_t @ W_ih^T + h @ W_hh^T     M=32, N=2048, K=256 (+K2=512)
    wmma_gemm_kernel<float><<<dim3((4 * DH) / 64, 1), 256, 0, stream>>>(
        embs_bf + (long long)t * DE, DT * DE, Wih_bf, DE, DE,
        hc_bf, DH + DENC, Whh_bf, DH, DH,
        nullptr, gates, (long long)4 * DH);

    lstm_cell_kernel<<<(DB * DH + 255) / 256, 256, 0, stream>>>(
        gates, b_ih, b_hh, cbuf, hbuf, hc_bf);

    attention_kernel<<<DB, 256, 0, stream>>>(hbuf, proj, enc_outputs, enc_lengths, hc_bf);

    // ho = [h|ctx] @ Wc^T + Wc_b           M=32, N=512, K=1024 -> bf16
    wmma_gemm_kernel<bf16_t><<<dim3(DH / 64, 1), 256, 0, stream>>>(
        hc_bf, DH + DENC, Wc_bf, DH + DENC, DH + DENC,
        nullptr, 0, nullptr, 0, 0, Wc_b, ho_bf, (long long)DH);

    // logits[:, t, :] = ho @ Wo^T + Wo_b   M=32, N=32000, K=512 (dominant GEMM)
    wmma_gemm_kernel<float><<<dim3(DV / 64, 1), 256, 0, stream>>>(
        ho_bf, DH, Wo_bf, DH, DH,
        nullptr, 0, nullptr, 0, 0, Wo_b,
        out + (long long)t * DV, (long long)DT * DV);
  }
}
